// STFT_60258391162989
// MI455X (gfx1250) — compile-verified, matrix operand order checked by reading, not run on hardware
//
#include <hip/hip_runtime.h>
#include <hip/hip_bf16.h>

typedef __attribute__((ext_vector_type(16))) __bf16 v16bf;
typedef __attribute__((ext_vector_type(8)))  __bf16 v8bf;
typedef __attribute__((ext_vector_type(8)))  float  v8f;

constexpr int kNfft  = 2048;
constexpr int kHop   = 512;
constexpr int kFreq  = 1025;     // N_FFT/2 + 1
constexpr int kLen   = 320000;   // samples per batch
constexpr int kBatch = 16;
constexpr int kT     = 626;      // (320000 + 2*1024 - 2048)/512 + 1
constexpr int MTILE  = 128;      // freq rows per workgroup
constexpr int NTILE  = 64;       // time cols per workgroup
constexpr int WIN    = (NTILE - 1) * kHop + kNfft;  // 34304 bf16 -> 68608 B LDS

constexpr size_t kWElems = (size_t)kFreq * kNfft;   // 2,099,200 per component
constexpr size_t kXElems = (size_t)kBatch * kLen;   // 5,120,000
constexpr size_t kWsNeed = (2 * kWElems + kXElems) * sizeof(__bf16);

union BfPack { v8bf h[2]; v16bf v; };

// Fragment loader for 16-bit 16x32 A (or 32x16 B) tiles, ISA 7.12.2 layout:
// lane&15 selects the row (A) / column (B); lanes<16 carry K=0..7,16..23 and
// lanes>=16 carry K=8..15,24..31 as two contiguous 8-element (16 B) runs.
__device__ inline v16bf load_frag_bf(const __bf16* p0, int lane) {
  const int h8 = (lane >> 4) << 3;                 // 0 or 8
  const __bf16* p = p0 + h8;
  BfPack r;
  r.h[0] = *reinterpret_cast<const v8bf*>(p);
  r.h[1] = *reinterpret_cast<const v8bf*>(p + 16);
  return r.v;
}

// f32 fragment loader with inline bf16 conversion (fallback path only).
__device__ inline v16bf load_a_frag_f32(const float* __restrict__ W, int row, int k0, int lane) {
  const int h8 = (lane >> 4) << 3;
  const float* p = W + (size_t)row * kNfft + (k0 + h8);
  const float4 x0 = *reinterpret_cast<const float4*>(p + 0);
  const float4 x1 = *reinterpret_cast<const float4*>(p + 4);
  const float4 x2 = *reinterpret_cast<const float4*>(p + 16);
  const float4 x3 = *reinterpret_cast<const float4*>(p + 20);
  v16bf a;
  a[0]  = (__bf16)x0.x; a[1]  = (__bf16)x0.y; a[2]  = (__bf16)x0.z; a[3]  = (__bf16)x0.w;
  a[4]  = (__bf16)x1.x; a[5]  = (__bf16)x1.y; a[6]  = (__bf16)x1.z; a[7]  = (__bf16)x1.w;
  a[8]  = (__bf16)x2.x; a[9]  = (__bf16)x2.y; a[10] = (__bf16)x2.z; a[11] = (__bf16)x2.w;
  a[12] = (__bf16)x3.x; a[13] = (__bf16)x3.y; a[14] = (__bf16)x3.z; a[15] = (__bf16)x3.w;
  return a;
}

// ---------------- Pre-pass: f32 -> bf16 bulk convert ----------------
__global__ __launch_bounds__(256) void cvt_f32_to_bf16(
    const float* __restrict__ src, __bf16* __restrict__ dst, int n) {
  int i = (blockIdx.x * 256 + threadIdx.x) * 8;
  if (i + 8 <= n) {
    const float4 a = *reinterpret_cast<const float4*>(src + i);
    const float4 b = *reinterpret_cast<const float4*>(src + i + 4);
    v8bf o;
    o[0] = (__bf16)a.x; o[1] = (__bf16)a.y; o[2] = (__bf16)a.z; o[3] = (__bf16)a.w;
    o[4] = (__bf16)b.x; o[5] = (__bf16)b.y; o[6] = (__bf16)b.z; o[7] = (__bf16)b.w;
    *reinterpret_cast<v8bf*>(dst + i) = o;
  } else {
    for (; i < n; ++i) dst[i] = (__bf16)src[i];
  }
}

// ---------------- Shared epilogue / accumulator core ----------------
template <typename AFragFn>
__device__ __forceinline__ void stft_core_store(
    v8f acc[2][2][2], float* __restrict__ out,
    int b, int f0, int t0, int wm, int wn, int lane) {
  const int m = lane & 15;
  const size_t plane = (size_t)kBatch * kFreq * kT;
  float* ob = out + (size_t)b * kFreq * kT;
  const int rhalf = (lane >> 4) << 3;
  #pragma unroll
  for (int c = 0; c < 2; ++c) {
    float* o = ob + (size_t)c * plane;
    #pragma unroll
    for (int mt = 0; mt < 2; ++mt) {
      #pragma unroll
      for (int nt = 0; nt < 2; ++nt) {
        const int col = t0 + wn * 32 + nt * 16 + m;
        if (col >= kT) continue;
        #pragma unroll
        for (int r = 0; r < 8; ++r) {
          const int row = f0 + wm * 32 + mt * 16 + rhalf + r;
          if (row < kFreq) o[(size_t)row * kT + col] = acc[c][mt][nt][r];
        }
      }
    }
  }
}

// ---------------- Fast path: all-bf16 operands from workspace ----------------
__global__ __launch_bounds__(256) void stft_wmma_bf16(
    const __bf16* __restrict__ xw, const __bf16* __restrict__ wrb,
    const __bf16* __restrict__ wib, float* __restrict__ out) {
  __shared__ __bf16 sx[WIN];

  const int b    = blockIdx.z;
  const int f0   = blockIdx.x * MTILE;
  const int t0   = blockIdx.y * NTILE;
  const int tid  = threadIdx.x;
  const int lane = tid & 31;
  const int wave = tid >> 5;
  const int wm   = wave >> 1;
  const int wn   = wave & 1;
  const int m    = lane & 15;

  // ---- Stage bf16 signal window into LDS ----
  const __bf16* xb = xw + (size_t)b * kLen;
  const int pstart = t0 * kHop - (kNfft / 2);   // window start in signal coords
  if (pstart >= 0 && pstart + WIN <= kLen) {
    // Interior block: pure byte copy -> CDNA5 async global->LDS (ASYNCcnt).
    const __bf16* src = xb + pstart;            // 16 B aligned (pstart mult. of 512)
    for (int i = tid * 8; i < WIN; i += 256 * 8) {
      unsigned lds_addr = (unsigned)(size_t)(&sx[i]);
      unsigned long long ga = (unsigned long long)(size_t)(src + i);
      asm volatile("global_load_async_to_lds_b128 %0, %1, off"
                   :: "v"(lds_addr), "v"(ga) : "memory");
    }
    asm volatile("s_wait_asynccnt 0" ::: "memory");
  } else {
    // Edge block: resolve reflect padding with a gather.
    for (int i = tid; i < WIN; i += 256) {
      int n = pstart + i;
      n = (n < 0) ? -n : n;
      if (n >= kLen) n = 2 * (kLen - 1) - n;
      sx[i] = xb[n];
    }
  }
  __syncthreads();

  v8f acc[2][2][2];
  #pragma unroll
  for (int c = 0; c < 2; ++c)
    #pragma unroll
    for (int i = 0; i < 2; ++i)
      #pragma unroll
      for (int j = 0; j < 2; ++j)
        acc[c][i][j] = (v8f){0.f, 0.f, 0.f, 0.f, 0.f, 0.f, 0.f, 0.f};

  int rowA0 = f0 + wm * 32 + m;       if (rowA0 > kFreq - 1) rowA0 = kFreq - 1;
  int rowA1 = f0 + wm * 32 + 16 + m;  if (rowA1 > kFreq - 1) rowA1 = kFreq - 1;
  const __bf16* wr0 = wrb + (size_t)rowA0 * kNfft;
  const __bf16* wr1 = wrb + (size_t)rowA1 * kNfft;
  const __bf16* wi0 = wib + (size_t)rowA0 * kNfft;
  const __bf16* wi1 = wib + (size_t)rowA1 * kNfft;
  const __bf16* bp0 = sx + (wn * 32 + m) * kHop;
  const __bf16* bp1 = bp0 + 16 * kHop;

  for (int k0 = 0; k0 < kNfft; k0 += 32) {
    const v16bf bfr0 = load_frag_bf(bp0 + k0, lane);
    const v16bf bfr1 = load_frag_bf(bp1 + k0, lane);
    const v16bf ar0  = load_frag_bf(wr0 + k0, lane);
    const v16bf ar1  = load_frag_bf(wr1 + k0, lane);
    const v16bf ai0  = load_frag_bf(wi0 + k0, lane);
    const v16bf ai1  = load_frag_bf(wi1 + k0, lane);

    acc[0][0][0] = __builtin_amdgcn_wmma_f32_16x16x32_bf16(false, ar0, false, bfr0, (short)0, acc[0][0][0], false, false);
    acc[0][0][1] = __builtin_amdgcn_wmma_f32_16x16x32_bf16(false, ar0, false, bfr1, (short)0, acc[0][0][1], false, false);
    acc[0][1][0] = __builtin_amdgcn_wmma_f32_16x16x32_bf16(false, ar1, false, bfr0, (short)0, acc[0][1][0], false, false);
    acc[0][1][1] = __builtin_amdgcn_wmma_f32_16x16x32_bf16(false, ar1, false, bfr1, (short)0, acc[0][1][1], false, false);
    acc[1][0][0] = __builtin_amdgcn_wmma_f32_16x16x32_bf16(false, ai0, false, bfr0, (short)0, acc[1][0][0], false, false);
    acc[1][0][1] = __builtin_amdgcn_wmma_f32_16x16x32_bf16(false, ai0, false, bfr1, (short)0, acc[1][0][1], false, false);
    acc[1][1][0] = __builtin_amdgcn_wmma_f32_16x16x32_bf16(false, ai1, false, bfr0, (short)0, acc[1][1][0], false, false);
    acc[1][1][1] = __builtin_amdgcn_wmma_f32_16x16x32_bf16(false, ai1, false, bfr1, (short)0, acc[1][1][1], false, false);
  }

  stft_core_store<int>(acc, out, b, f0, t0, wm, wn, lane);
}

// ---------------- Fallback: self-contained (converts inline) ----------------
__global__ __launch_bounds__(256) void stft_wmma_f32in(
    const float* __restrict__ x, const float* __restrict__ wr,
    const float* __restrict__ wi, float* __restrict__ out) {
  __shared__ __bf16 sx[WIN];

  const int b    = blockIdx.z;
  const int f0   = blockIdx.x * MTILE;
  const int t0   = blockIdx.y * NTILE;
  const int tid  = threadIdx.x;
  const int lane = tid & 31;
  const int wave = tid >> 5;
  const int wm   = wave >> 1;
  const int wn   = wave & 1;
  const int m    = lane & 15;

  const float* xb = x + (size_t)b * kLen;
  const int pstart = t0 * kHop - (kNfft / 2);
  for (int i = tid; i < WIN; i += 256) {
    int n = pstart + i;
    n = (n < 0) ? -n : n;
    if (n >= kLen) n = 2 * (kLen - 1) - n;
    sx[i] = (__bf16)xb[n];
  }
  __syncthreads();

  v8f acc[2][2][2];
  #pragma unroll
  for (int c = 0; c < 2; ++c)
    #pragma unroll
    for (int i = 0; i < 2; ++i)
      #pragma unroll
      for (int j = 0; j < 2; ++j)
        acc[c][i][j] = (v8f){0.f, 0.f, 0.f, 0.f, 0.f, 0.f, 0.f, 0.f};

  int rowA0 = f0 + wm * 32 + m;       if (rowA0 > kFreq - 1) rowA0 = kFreq - 1;
  int rowA1 = f0 + wm * 32 + 16 + m;  if (rowA1 > kFreq - 1) rowA1 = kFreq - 1;
  const __bf16* bp0 = sx + (wn * 32 + m) * kHop;
  const __bf16* bp1 = bp0 + 16 * kHop;

  for (int k0 = 0; k0 < kNfft; k0 += 32) {
    const v16bf bfr0 = load_frag_bf(bp0 + k0, lane);
    const v16bf bfr1 = load_frag_bf(bp1 + k0, lane);
    const v16bf ar0  = load_a_frag_f32(wr, rowA0, k0, lane);
    const v16bf ar1  = load_a_frag_f32(wr, rowA1, k0, lane);
    const v16bf ai0  = load_a_frag_f32(wi, rowA0, k0, lane);
    const v16bf ai1  = load_a_frag_f32(wi, rowA1, k0, lane);

    acc[0][0][0] = __builtin_amdgcn_wmma_f32_16x16x32_bf16(false, ar0, false, bfr0, (short)0, acc[0][0][0], false, false);
    acc[0][0][1] = __builtin_amdgcn_wmma_f32_16x16x32_bf16(false, ar0, false, bfr1, (short)0, acc[0][0][1], false, false);
    acc[0][1][0] = __builtin_amdgcn_wmma_f32_16x16x32_bf16(false, ar1, false, bfr0, (short)0, acc[0][1][0], false, false);
    acc[0][1][1] = __builtin_amdgcn_wmma_f32_16x16x32_bf16(false, ar1, false, bfr1, (short)0, acc[0][1][1], false, false);
    acc[1][0][0] = __builtin_amdgcn_wmma_f32_16x16x32_bf16(false, ai0, false, bfr0, (short)0, acc[1][0][0], false, false);
    acc[1][0][1] = __builtin_amdgcn_wmma_f32_16x16x32_bf16(false, ai0, false, bfr1, (short)0, acc[1][0][1], false, false);
    acc[1][1][0] = __builtin_amdgcn_wmma_f32_16x16x32_bf16(false, ai1, false, bfr0, (short)0, acc[1][1][0], false, false);
    acc[1][1][1] = __builtin_amdgcn_wmma_f32_16x16x32_bf16(false, ai1, false, bfr1, (short)0, acc[1][1][1], false, false);
  }

  stft_core_store<int>(acc, out, b, f0, t0, wm, wn, lane);
}

extern "C" void kernel_launch(void* const* d_in, const int* in_sizes, int n_in,
                              void* d_out, int out_size, void* d_ws, size_t ws_size,
                              hipStream_t stream) {
  const float* x  = (const float*)d_in[0];
  const float* wr = (const float*)d_in[1];
  const float* wi = (const float*)d_in[2];
  float* out = (float*)d_out;

  dim3 grid((kFreq + MTILE - 1) / MTILE,   // 9
            (kT + NTILE - 1) / NTILE,      // 10
            kBatch);                        // 16

  if (ws_size >= kWsNeed) {
    __bf16* wrb = (__bf16*)d_ws;
    __bf16* wib = wrb + kWElems;
    __bf16* xw  = wib + kWElems;
    const int nW = (int)kWElems, nX = (int)kXElems;
    cvt_f32_to_bf16<<<(nW / 8 + 255) / 256, 256, 0, stream>>>(wr, wrb, nW);
    cvt_f32_to_bf16<<<(nW / 8 + 255) / 256, 256, 0, stream>>>(wi, wib, nW);
    cvt_f32_to_bf16<<<(nX / 8 + 255) / 256, 256, 0, stream>>>(x,  xw,  nX);
    stft_wmma_bf16<<<grid, dim3(256), 0, stream>>>(xw, wrb, wib, out);
  } else {
    stft_wmma_f32in<<<grid, dim3(256), 0, stream>>>(x, wr, wi, out);
  }
}
